// MambaBlock_26036091748689
// MI455X (gfx1250) — compile-verified, matrix operand order checked by reading, not run on hardware
//
#include <hip/hip_runtime.h>
#include <hip/hip_bf16.h>
#include <math.h>

// ---------------------------------------------------------------------------
// Mamba block for MI455X (gfx1250).
//   * Large GEMMs: bf16 WMMA (v_wmma_f32_16x16x32_bf16, f32 accum).
//     Wave tile 32x64 (8 WMMAs / K-step), block tile 64x256 (8 waves),
//     fragments loaded as two contiguous global_load_b128 per operand row
//     (A row-major, B pre-transposed to N-major/K-contiguous bf16).
//     __launch_bounds__(256,2) so the 8 accumulator tiles + fragments live
//     entirely in VGPRs (2 waves/SIMD -> ~double default VGPR budget).
//   * Selective scan: 1 thread per (batch, d_inner) channel, h[16] in regs,
//     per-step B/C staged through LDS.
// ---------------------------------------------------------------------------

#define DMODEL 1024
#define DSTATE 16
#define DCONV  4
#define DINNER 2048
#define BATCH  2
#define SEQ    2048
#define MROWS  (BATCH * SEQ)     // 4096

typedef __attribute__((ext_vector_type(16))) __bf16 v16bf;
typedef __attribute__((ext_vector_type(8)))  __bf16 v8bf;
typedef __attribute__((ext_vector_type(8)))  float  v8f;

__device__ __forceinline__ __bf16 f2bf(float x) {
    union { float f; unsigned u; } c; c.f = x;
    unsigned r = c.u + 0x7FFFu + ((c.u >> 16) & 1u);   // round-to-nearest-even
    unsigned short h = (unsigned short)(r >> 16);
    union { unsigned short s; __bf16 b; } o; o.s = h;
    return o.b;
}

__device__ __forceinline__ float silu_f(float x) {
    return x / (1.0f + __expf(-x));
}

// ---------------------------------------------------------------------------
// 1) LayerCentering (subtract feature mean) + convert to bf16.
// ---------------------------------------------------------------------------
__global__ void center_bf16_kernel(const float* __restrict__ inp,
                                   __bf16* __restrict__ xin_bf) {
    __shared__ float red[256];
    const int m   = blockIdx.x;             // 0..MROWS-1
    const int tid = threadIdx.x;            // 0..255
    const size_t base = (size_t)m * DMODEL;

    float s = 0.0f;
    #pragma unroll
    for (int i = 0; i < DMODEL / 256; ++i) s += inp[base + tid + i * 256];
    red[tid] = s;
    __syncthreads();
    for (int w = 128; w > 0; w >>= 1) {
        if (tid < w) red[tid] += red[tid + w];
        __syncthreads();
    }
    const float mean = red[0] * (1.0f / DMODEL);
    #pragma unroll
    for (int i = 0; i < DMODEL / 256; ++i) {
        int idx = tid + i * 256;
        xin_bf[base + idx] = f2bf(inp[base + idx] - mean);
    }
}

// ---------------------------------------------------------------------------
// 2) Weight convert+transpose: W (K x N, f32) -> Wt (N x K, bf16).
// ---------------------------------------------------------------------------
__global__ void wtrans_bf16_kernel(const float* __restrict__ W,
                                   __bf16* __restrict__ Wt, int K, int N) {
    size_t idx = (size_t)blockIdx.x * blockDim.x + threadIdx.x;
    if (idx >= (size_t)K * N) return;
    int k = (int)(idx / N);
    int n = (int)(idx % N);
    Wt[(size_t)n * K + k] = f2bf(W[idx]);
}

// ---------------------------------------------------------------------------
// 3) bf16 WMMA GEMM: C(f32, MxN) = A(bf16, MxK row-major) * Bt(bf16, NxK).
//    Block = 256 threads (8 waves, 2 down M x 4 across N), block tile 64x256,
//    wave tile 32x64.  Requires M%64==0, N%256==0, K%32==0 (true here).
// ---------------------------------------------------------------------------
__device__ __forceinline__ v16bf load_frag(const __bf16* __restrict__ p) {
    v8bf lo = *(const v8bf*)(p);        // K = kk + half*8 .. +7
    v8bf hi = *(const v8bf*)(p + 16);   // K = kk + 16 + half*8 .. +7
    return __builtin_shufflevector(lo, hi, 0, 1, 2, 3, 4, 5, 6, 7,
                                           8, 9, 10, 11, 12, 13, 14, 15);
}

__device__ __forceinline__ v8f wmma_bf16(v16bf a, v16bf b, v8f c) {
    return __builtin_amdgcn_wmma_f32_16x16x32_bf16(
        /*neg_a=*/false, a, /*neg_b=*/false, b,
        /*c_mod=*/(short)0, c, /*reuse_a=*/false, /*reuse_b=*/false);
}

__global__ void __launch_bounds__(256, 2)
gemm_bf16_wmma_kernel(const __bf16* __restrict__ A,
                      const __bf16* __restrict__ Bt,
                      float* __restrict__ C,
                      int M, int N, int K) {
    const int lane  = threadIdx.x & 31;
    const int wave  = threadIdx.x >> 5;      // 0..7
    const int waveM = wave & 1;              // 2 waves down M
    const int waveN = wave >> 1;             // 4 waves across N
    const int r     = lane & 15;
    const int half  = lane >> 4;
    const int hofs  = half * 8;

    const int rowBase = blockIdx.y * 64  + waveM * 32;
    const int colBase = blockIdx.x * 256 + waveN * 64;

    const __bf16* aRow0 = A  + (size_t)(rowBase + r)      * K + hofs;
    const __bf16* aRow1 = A  + (size_t)(rowBase + 16 + r) * K + hofs;
    const __bf16* bRow0 = Bt + (size_t)(colBase + r)      * K + hofs;
    const __bf16* bRow1 = Bt + (size_t)(colBase + 16 + r) * K + hofs;
    const __bf16* bRow2 = Bt + (size_t)(colBase + 32 + r) * K + hofs;
    const __bf16* bRow3 = Bt + (size_t)(colBase + 48 + r) * K + hofs;

    v8f acc[2][4] = {};

    #pragma unroll 1
    for (int kk = 0; kk < K; kk += 32) {
        if (kk + 128 < K) {   // GL2-temporal prefetch (global_prefetch_b8)
            __builtin_prefetch(aRow0 + kk + 128, 0, 3);
            __builtin_prefetch(bRow0 + kk + 128, 0, 3);
        }
        v16bf a0 = load_frag(aRow0 + kk);
        v16bf a1 = load_frag(aRow1 + kk);
        v16bf b0 = load_frag(bRow0 + kk);
        v16bf b1 = load_frag(bRow1 + kk);
        v16bf b2 = load_frag(bRow2 + kk);
        v16bf b3 = load_frag(bRow3 + kk);
        acc[0][0] = wmma_bf16(a0, b0, acc[0][0]);
        acc[0][1] = wmma_bf16(a0, b1, acc[0][1]);
        acc[0][2] = wmma_bf16(a0, b2, acc[0][2]);
        acc[0][3] = wmma_bf16(a0, b3, acc[0][3]);
        acc[1][0] = wmma_bf16(a1, b0, acc[1][0]);
        acc[1][1] = wmma_bf16(a1, b1, acc[1][1]);
        acc[1][2] = wmma_bf16(a1, b2, acc[1][2]);
        acc[1][3] = wmma_bf16(a1, b3, acc[1][3]);
    }

    // C/D layout: VGPR v -> M = v + 8*half, N = lane%16.
    const int ldc = N;
    #pragma unroll
    for (int tm = 0; tm < 2; ++tm) {
        #pragma unroll
        for (int tn = 0; tn < 4; ++tn) {
            const int mB = rowBase + tm * 16 + half * 8;
            const int nB = colBase + tn * 16 + r;
            #pragma unroll
            for (int v = 0; v < 8; ++v)
                C[(size_t)(mB + v) * ldc + nB] = acc[tm][tn][v];
        }
    }
}

// ---------------------------------------------------------------------------
// 4) Depthwise conv1d (k=4, pad (1,2)) + bias + SiLU on the x-half of xz.
// ---------------------------------------------------------------------------
__global__ void conv_silu_kernel(const float* __restrict__ xz,
                                 const float* __restrict__ conv_w,
                                 const float* __restrict__ conv_b,
                                 float* __restrict__ xc,
                                 __bf16* __restrict__ xc_bf) {
    size_t idx = (size_t)blockIdx.x * blockDim.x + threadIdx.x;
    if (idx >= (size_t)MROWS * DINNER) return;
    int m = (int)(idx / DINNER);     // b*SEQ + l
    int c = (int)(idx % DINNER);
    int b = m / SEQ;
    int l = m % SEQ;

    float acc = conv_b[c];
    #pragma unroll
    for (int j = 0; j < DCONV; ++j) {
        int lj = l - 1 + j;
        if (lj >= 0 && lj < SEQ)
            acc += conv_w[j * DINNER + c] *
                   xz[((size_t)(b * SEQ + lj)) * (2 * DINNER) + c];
    }
    float v = silu_f(acc);
    xc[idx]    = v;
    xc_bf[idx] = f2bf(v);
}

// ---------------------------------------------------------------------------
// 5) delta = softplus(delta_pre + b_dt)
// ---------------------------------------------------------------------------
__global__ void softplus_kernel(float* __restrict__ delta,
                                const float* __restrict__ b_dt) {
    size_t idx = (size_t)blockIdx.x * blockDim.x + threadIdx.x;
    if (idx >= (size_t)MROWS * DINNER) return;
    int n = (int)(idx % DINNER);
    float x = delta[idx] + b_dt[n];
    delta[idx] = (x > 20.0f) ? x : log1pf(__expf(x));
}

// ---------------------------------------------------------------------------
// 6) x_db = x_conv @ W_x (N = 32; one wave per row: A broadcast, B coalesced)
// ---------------------------------------------------------------------------
__global__ void xdb_kernel(const float* __restrict__ xc,
                           const float* __restrict__ W_x,
                           float* __restrict__ xdb) {
    size_t idx = (size_t)blockIdx.x * blockDim.x + threadIdx.x;
    if (idx >= (size_t)MROWS * 2 * DSTATE) return;
    int m = (int)(idx / (2 * DSTATE));
    int n = (int)(idx % (2 * DSTATE));
    const float* a = xc + (size_t)m * DINNER;
    float acc = 0.0f;
    for (int k = 0; k < DINNER; ++k) acc += a[k] * W_x[(size_t)k * (2 * DSTATE) + n];
    xdb[idx] = acc;
}

// ---------------------------------------------------------------------------
// 7) Selective scan.  grid = (DINNER/256, BATCH); 1 thread = 1 channel.
// ---------------------------------------------------------------------------
__global__ void scan_kernel(const float* __restrict__ delta,
                            const float* __restrict__ u,
                            const float* __restrict__ xdb,
                            const float* __restrict__ Amat,
                            float* __restrict__ y) {
    __shared__ float sBC[2 * DSTATE];
    const int c = blockIdx.x * 256 + threadIdx.x;   // 0..DINNER-1
    const int b = blockIdx.y;

    float h[DSTATE];
    float Ar[DSTATE];
    #pragma unroll
    for (int s = 0; s < DSTATE; ++s) { h[s] = 0.0f; Ar[s] = Amat[(size_t)c * DSTATE + s]; }

    for (int l = 0; l < SEQ; ++l) {
        const size_t row = (size_t)(b * SEQ + l);
        if (threadIdx.x < 2 * DSTATE)
            sBC[threadIdx.x] = xdb[row * (2 * DSTATE) + threadIdx.x];
        __syncthreads();

        const float dt = delta[row * DINNER + c];
        const float ut = u[row * DINNER + c];
        const float du = dt * ut;
        float yv = 0.0f;
        #pragma unroll
        for (int s = 0; s < DSTATE; ++s) {
            float dA = __expf(dt * Ar[s]);
            h[s] = dA * h[s] + du * sBC[s];
            yv  += h[s] * sBC[DSTATE + s];
        }
        y[row * DINNER + c] = yv;
        __syncthreads();
    }
}

// ---------------------------------------------------------------------------
// 8) Gate: y = (y + x*D) * silu(z), written as bf16.
// ---------------------------------------------------------------------------
__global__ void gate_kernel(const float* __restrict__ y,
                            const float* __restrict__ xc,
                            const float* __restrict__ xz,
                            const float* __restrict__ Dvec,
                            __bf16* __restrict__ y_bf) {
    size_t idx = (size_t)blockIdx.x * blockDim.x + threadIdx.x;
    if (idx >= (size_t)MROWS * DINNER) return;
    int m = (int)(idx / DINNER);
    int c = (int)(idx % DINNER);
    float z  = xz[(size_t)m * (2 * DINNER) + DINNER + c];
    float yv = (y[idx] + xc[idx] * Dvec[c]) * silu_f(z);
    y_bf[idx] = f2bf(yv);
}

// ---------------------------------------------------------------------------
// 9) Residual add in place on the GEMM output.
// ---------------------------------------------------------------------------
__global__ void residual_kernel(float* __restrict__ out,
                                const float* __restrict__ inp) {
    size_t idx = (size_t)blockIdx.x * blockDim.x + threadIdx.x;
    if (idx >= (size_t)MROWS * DMODEL) return;
    out[idx] += inp[idx];
}

// ---------------------------------------------------------------------------
// Launch
// ---------------------------------------------------------------------------
extern "C" void kernel_launch(void* const* d_in, const int* in_sizes, int n_in,
                              void* d_out, int out_size, void* d_ws, size_t ws_size,
                              hipStream_t stream) {
    (void)in_sizes; (void)n_in; (void)out_size; (void)ws_size;
    const float* inp    = (const float*)d_in[0];
    const float* W_in   = (const float*)d_in[1];
    const float* conv_w = (const float*)d_in[2];
    const float* conv_b = (const float*)d_in[3];
    const float* W_x    = (const float*)d_in[4];
    const float* W_dt   = (const float*)d_in[5];
    const float* b_dt   = (const float*)d_in[6];
    const float* W_out  = (const float*)d_in[7];
    const float* Amat   = (const float*)d_in[8];
    const float* Dvec   = (const float*)d_in[9];
    float* out = (float*)d_out;

    char* base = (char*)d_ws;
    size_t off = 0;
    auto wsalloc = [&](size_t bytes) -> void* {
        void* p = base + off;
        off = (off + bytes + 255) & ~(size_t)255;
        return p;
    };

    __bf16* xin_bf = (__bf16*)wsalloc((size_t)MROWS * DMODEL * 2);          //  8 MB
    __bf16* WinT   = (__bf16*)wsalloc((size_t)2 * DINNER * DMODEL * 2);     //  8 MB
    float*  xz     = (float*) wsalloc((size_t)MROWS * 2 * DINNER * 4);      // 64 MB
    float*  xc     = (float*) wsalloc((size_t)MROWS * DINNER * 4);          // 32 MB
    __bf16* xc_bf  = (__bf16*)wsalloc((size_t)MROWS * DINNER * 2);          // 16 MB
    __bf16* WdtT   = (__bf16*)wsalloc((size_t)DINNER * DINNER * 2);         //  8 MB
    float*  delta  = (float*) wsalloc((size_t)MROWS * DINNER * 4);          // 32 MB
    float*  xdb    = (float*) wsalloc((size_t)MROWS * 2 * DSTATE * 4);      // .5 MB
    float*  ybuf   = (float*) wsalloc((size_t)MROWS * DINNER * 4);          // 32 MB
    __bf16* y_bf   = (__bf16*)wsalloc((size_t)MROWS * DINNER * 2);          // 16 MB
    __bf16* WoutT  = (__bf16*)wsalloc((size_t)DMODEL * DINNER * 2);         //  4 MB

    // 1) center + bf16
    center_bf16_kernel<<<MROWS, 256, 0, stream>>>(inp, xin_bf);

    // 2) weight transposes (f32 -> bf16, N-major)
    {
        size_t n1 = (size_t)DMODEL * 2 * DINNER;
        wtrans_bf16_kernel<<<(unsigned)((n1 + 255) / 256), 256, 0, stream>>>(W_in, WinT, DMODEL, 2 * DINNER);
        size_t n2 = (size_t)DINNER * DINNER;
        wtrans_bf16_kernel<<<(unsigned)((n2 + 255) / 256), 256, 0, stream>>>(W_dt, WdtT, DINNER, DINNER);
        size_t n3 = (size_t)DINNER * DMODEL;
        wtrans_bf16_kernel<<<(unsigned)((n3 + 255) / 256), 256, 0, stream>>>(W_out, WoutT, DINNER, DMODEL);
    }

    // 3) xz = x_in @ W_in : M=4096, N=4096, K=1024
    {
        dim3 g((2 * DINNER) / 256, MROWS / 64);
        gemm_bf16_wmma_kernel<<<g, 256, 0, stream>>>(xin_bf, WinT, xz, MROWS, 2 * DINNER, DMODEL);
    }

    // 4) depthwise conv + SiLU
    {
        size_t n = (size_t)MROWS * DINNER;
        conv_silu_kernel<<<(unsigned)((n + 255) / 256), 256, 0, stream>>>(xz, conv_w, conv_b, xc, xc_bf);
    }

    // 5) delta_pre = x @ W_dt : M=4096, N=2048, K=2048 ; then softplus(+b_dt)
    {
        dim3 g(DINNER / 256, MROWS / 64);
        gemm_bf16_wmma_kernel<<<g, 256, 0, stream>>>(xc_bf, WdtT, delta, MROWS, DINNER, DINNER);
        size_t n = (size_t)MROWS * DINNER;
        softplus_kernel<<<(unsigned)((n + 255) / 256), 256, 0, stream>>>(delta, b_dt);
    }

    // 6) x_db = x @ W_x (skinny N=32)
    {
        size_t n = (size_t)MROWS * 2 * DSTATE;
        xdb_kernel<<<(unsigned)((n + 255) / 256), 256, 0, stream>>>(xc, W_x, xdb);
    }

    // 7) selective scan
    {
        dim3 g(DINNER / 256, BATCH);
        scan_kernel<<<g, 256, 0, stream>>>(delta, xc, xdb, Amat, ybuf);
    }

    // 8) gate
    {
        size_t n = (size_t)MROWS * DINNER;
        gate_kernel<<<(unsigned)((n + 255) / 256), 256, 0, stream>>>(ybuf, xc, xz, Dvec, y_bf);
    }

    // 9) out = y @ W_out : M=4096, N=1024, K=2048, written straight to d_out
    {
        dim3 g(DMODEL / 256, MROWS / 64);
        gemm_bf16_wmma_kernel<<<g, 256, 0, stream>>>(y_bf, WoutT, out, MROWS, DMODEL, DINNER);
    }

    // 10) residual
    {
        size_t n = (size_t)MROWS * DMODEL;
        residual_kernel<<<(unsigned)((n + 255) / 256), 256, 0, stream>>>(out, inp);
    }
}
